// ModelGcnHop_56324201120151
// MI455X (gfx1250) — matrix operand order repaired and checked by reading, not compiled, not run on hardware
//
#include <hip/hip_runtime.h>
#include <hip/hip_bf16.h>

#define E 128
#define S 512
#define BSUB 192
#define NROWS (BSUB * S)

typedef _Float16 half16v __attribute__((ext_vector_type(16)));
typedef _Float16 half8v  __attribute__((ext_vector_type(8)));
typedef float    float8v __attribute__((ext_vector_type(8)));

// ---------------- wave32 reductions ----------------
__device__ __forceinline__ float wave_sum(float v) {
#pragma unroll
  for (int m = 16; m >= 1; m >>= 1) v += __shfl_xor(v, m, 32);
  return v;
}
__device__ __forceinline__ float wave_max(float v) {
#pragma unroll
  for (int m = 16; m >= 1; m >>= 1) v = fmaxf(v, __shfl_xor(v, m, 32));
  return v;
}

// ---------------- CDNA5 async global->LDS copy (ASYNCcnt) ----------------
// GV mode: vdst = LDS byte address (low 32 bits of generic shared ptr per the
// flat-aperture rule); vaddr = 64-bit global address.
__device__ __forceinline__ void async_copy_b128(void* lds_dst, const void* gsrc) {
  unsigned int loff = (unsigned int)(uintptr_t)lds_dst;
  unsigned long long ga = (unsigned long long)(uintptr_t)gsrc;
  asm volatile("global_load_async_to_lds_b128 %0, %1, off"
               :: "v"(loff), "v"(ga) : "memory");
}
__device__ __forceinline__ void async_wait_all() {
  asm volatile("s_wait_asynccnt 0x0" ::: "memory");
}

// ---------------- WMMA fragment loaders (CDNA5 wave32 layouts) ----------------
// A (16x32 f16): lane l holds row m=l&15; elems 0..7 -> K=e0+(l>>4)*8..+7,
// elems 8..15 -> K=e0+16+(l>>4)*8..+7  (two contiguous 16B loads)
__device__ __forceinline__ half16v load_frag_A(const _Float16* base, int lane,
                                               int rowStride, int e0) {
  const int m = lane & 15, g = lane >> 4;
  const _Float16* p = base + m * rowStride + e0 + g * 8;
  half8v lo = *(const half8v*)p;
  half8v hi = *(const half8v*)(p + 16);
  half16v r;
#pragma unroll
  for (int i = 0; i < 8; ++i) { r[i] = lo[i]; r[i + 8] = hi[i]; }
  return r;
}
// B (32x16 f16): lane l holds col n=l&15; elems 0..7 -> K=e0+(l>>4)*8..+7,
// elems 8..15 -> K=e0+16+(l>>4)*8..+7. Row-major source [n][K] (two 16B loads).
__device__ __forceinline__ half16v load_frag_B(const _Float16* base, int lane,
                                               int rowStride, int e0) {
  const int n = lane & 15, g = lane >> 4;
  const _Float16* p = base + n * rowStride + e0 + g * 8;
  half8v lo = *(const half8v*)p;
  half8v hi = *(const half8v*)(p + 16);
  half16v r;
#pragma unroll
  for (int i = 0; i < 8; ++i) { r[i] = lo[i]; r[i + 8] = hi[i]; }
  return r;
}
__device__ __forceinline__ float8v wmma_f16(half16v a, half16v b, float8v c) {
  return __builtin_amdgcn_wmma_f32_16x16x32_f16(false, a, false, b, (short)0, c,
                                                false, false);
}

// ============ Kernel 0: one-shot f32 -> f16 weight conversion ============
__global__ void __launch_bounds__(256)
wcvt_kernel(const float* __restrict__ Wq, const float* __restrict__ Wk,
            const float* __restrict__ Wv, const float* __restrict__ Wo,
            _Float16* __restrict__ wh) {
  const int i = blockIdx.x * 256 + threadIdx.x;  // 4*E*E total
  const float* srcs[4] = {Wq, Wk, Wv, Wo};
  wh[i] = (_Float16)srcs[i >> 14][i & (E * E - 1)];
}

// ============ Kernel 1: LN + ReLU + QKV (q,k row-major f16; v transposed) ============
__global__ void __launch_bounds__(256)
qkv_kernel(const float* __restrict__ x, const _Float16* __restrict__ wh,
           const float* __restrict__ bq, const float* __restrict__ bk,
           const float* __restrict__ bv,
           _Float16* __restrict__ qh, _Float16* __restrict__ kh,
           _Float16* __restrict__ vT) {
  extern __shared__ char dyn_smem[];
  _Float16* sW = (_Float16*)dyn_smem;      // 3 * 128*128 f16 (Wq,Wk,Wv)
  _Float16* sX = sW + 3 * E * E;           // 128*128 f16 (LN+ReLU activations)
  const int tid = threadIdx.x, lane = tid & 31, wave = tid >> 5;
  const int rowBase = blockIdx.x * 128;

  // async-DMA the f16 weights into LDS; overlaps with the LN pass below
  for (int i = tid; i < (3 * E * E) / 8; i += 256)
    async_copy_b128(sW + i * 8, wh + i * 8);

  // LayerNorm + ReLU: each wave handles 16 rows cooperatively (fp32 stats)
  for (int r = 0; r < 16; ++r) {
    const int row = rowBase + wave * 16 + r;
    float4 v = ((const float4*)(x + (size_t)row * E))[lane];
    float s  = wave_sum(v.x + v.y + v.z + v.w);
    float s2 = wave_sum(v.x * v.x + v.y * v.y + v.z * v.z + v.w * v.w);
    float mean = s * (1.0f / E);
    float rstd = rsqrtf(s2 * (1.0f / E) - mean * mean + 1e-5f);
    _Float16* dst = sX + (wave * 16 + r) * E + lane * 4;
    dst[0] = (_Float16)fmaxf((v.x - mean) * rstd, 0.0f);
    dst[1] = (_Float16)fmaxf((v.y - mean) * rstd, 0.0f);
    dst[2] = (_Float16)fmaxf((v.z - mean) * rstd, 0.0f);
    dst[3] = (_Float16)fmaxf((v.w - mean) * rstd, 0.0f);
  }
  async_wait_all();
  __syncthreads();

  half16v A[4];
#pragma unroll
  for (int kk = 0; kk < 4; ++kk)
    A[kk] = load_frag_A(sX + wave * 16 * E, lane, E, kk * 32);

  const int n = lane & 15, ghi = lane >> 4;
  const int b = rowBase / S, srow0 = (rowBase % S) + wave * 16;
  const float* bias[3] = {bq, bk, bv};
  _Float16* qkdst[2] = {qh, kh};

#pragma unroll
  for (int p = 0; p < 3; ++p) {  // fully unrolled: straight-line 96-WMMA body
    const _Float16* W = sW + p * E * E;  // row-major [f][e]; B col n = W row f
#pragma unroll
    for (int fb = 0; fb < 8; ++fb) {
      const float bb = bias[p][fb * 16 + n];
      half16v Bf[4];
#pragma unroll
      for (int kk = 0; kk < 4; ++kk)
        Bf[kk] = load_frag_B(W + fb * 16 * E, lane, E, kk * 32);
      float8v acc = {};
#pragma unroll
      for (int kk = 0; kk < 4; ++kk) acc = wmma_f16(A[kk], Bf[kk], acc);
      if (p < 2) {
        _Float16* dst = qkdst[p] +
                        ((size_t)rowBase + wave * 16) * E + fb * 16 + n;
#pragma unroll
        for (int g = 0; g < 8; ++g)
          dst[(size_t)(g + 8 * ghi) * E] = (_Float16)(acc[g] + bb);
      } else {  // v stored transposed: vT[b][f][srow]
        _Float16* dst = vT + ((size_t)b * E + fb * 16 + n) * S + srow0;
#pragma unroll
        for (int g = 0; g < 8; ++g)
          dst[g + 8 * ghi] = (_Float16)(acc[g] + bb);
      }
    }
  }
}

// ============ Kernel 2: masked attention, one wave per 16-query tile ============
#define SC_STRIDE 516   // fp32 score row stride (pad vs 512)
#define AT_STRIDE 528   // f16 prob row stride, overlaid on the score buffer:
// prob row r spans bytes [1056r, 1056r+1024) < score row r+1 at 2064(r+1),
// and each score row is fully register-resident before being overwritten.
__global__ void __launch_bounds__(32)
attn_kernel(const float* __restrict__ adj, const _Float16* __restrict__ qh,
            const _Float16* __restrict__ kh, const _Float16* __restrict__ vT,
            float* __restrict__ oat) {
  extern __shared__ char dyn_smem[];
  float*    sScore = (float*)dyn_smem;            // [16][516] fp32
  _Float16* sAttn  = (_Float16*)dyn_smem;         // [16][528] f16 (overlay)
  float*    sRinv  = (float*)(sScore + 16 * SC_STRIDE);  // [16]
  const int lane = threadIdx.x & 31;
  const int b  = blockIdx.x >> 5;   // 32 query tiles per subgraph
  const int q0 = (blockIdx.x & 31) * 16;
  const int n = lane & 15, ghi = lane >> 4;

  // ---- scores = q @ k^T  (K/V stream through the 192MB L2) ----
  const _Float16* qbase = qh + ((size_t)b * S + q0) * E;
  half16v Aq[4];
#pragma unroll
  for (int kk = 0; kk < 4; ++kk) Aq[kk] = load_frag_A(qbase, lane, E, kk * 32);

  const _Float16* kbase = kh + (size_t)b * S * E;
  for (int nb = 0; nb < 32; ++nb) {
    half16v Bk[4];
#pragma unroll
    for (int kk = 0; kk < 4; ++kk)
      Bk[kk] = load_frag_B(kbase + nb * 16 * E, lane, E, kk * 32);
    float8v acc = {};
#pragma unroll
    for (int kk = 0; kk < 4; ++kk) acc = wmma_f16(Aq[kk], Bk[kk], acc);
#pragma unroll
    for (int g = 0; g < 8; ++g)
      sScore[(g + 8 * ghi) * SC_STRIDE + nb * 16 + n] = acc[g];
  }
  __syncthreads();

  // ---- masked softmax (deg-scaled). 1/sqrt(deg) > 0 factors out of the max. ----
  for (int r = 0; r < 16; ++r) {
    const float* adjRow = adj + ((size_t)b * S + q0 + r) * S;
    if (r < 15) __builtin_prefetch(adjRow + S + lane * 16, 0, 0);
    float av[16], sv[16];
    float deg = 0.0f, mx = -1e30f;
#pragma unroll
    for (int i = 0; i < 16; ++i) {
      const int kk = i * 32 + lane;
      av[i] = adjRow[kk];
      sv[i] = sScore[r * SC_STRIDE + kk];
      deg += av[i];
      mx = (av[i] != 0.0f) ? fmaxf(mx, sv[i]) : mx;
    }
    deg = wave_sum(deg);
    mx  = wave_max(mx);
    const float rdeg = rsqrtf(deg);   // deg >= 1 (self loops)
    const float m = mx * rdeg;
    float ssum = 0.0f;
#pragma unroll
    for (int i = 0; i < 16; ++i) {
      const int kk = i * 32 + lane;
      const float pr = (av[i] != 0.0f) ? __expf(sv[i] * rdeg - m) : 0.0f;
      ssum += pr;
      sAttn[r * AT_STRIDE + kk] = (_Float16)pr;  // overlays consumed scores
    }
    ssum = wave_sum(ssum);
    if (lane == 0) sRinv[r] = 1.0f / ssum;
  }
  __syncthreads();

  // ---- out = attn @ v   (A from LDS probs, B from transposed V via L2) ----
  const _Float16* vtb = vT + (size_t)b * E * S;
  float8v acc[8];
#pragma unroll
  for (int fb = 0; fb < 8; ++fb) acc[fb] = (float8v){};
  for (int kb = 0; kb < 16; ++kb) {
    half16v Aa = load_frag_A(sAttn, lane, AT_STRIDE, kb * 32);
    half16v Bv[8];
#pragma unroll
    for (int fb = 0; fb < 8; ++fb)
      Bv[fb] = load_frag_B(vtb + fb * 16 * S, lane, S, kb * 32);
#pragma unroll
    for (int fb = 0; fb < 8; ++fb) acc[fb] = wmma_f16(Aa, Bv[fb], acc[fb]);
  }
  float* obase = oat + ((size_t)b * S + q0) * E;
#pragma unroll
  for (int fb = 0; fb < 8; ++fb) {
#pragma unroll
    for (int g = 0; g < 8; ++g) {
      const int m2 = g + 8 * ghi;
      obase[(size_t)m2 * E + fb * 16 + n] = acc[fb][g] * sRinv[m2];
    }
  }
}

// ============ Kernel 3: LN + ReLU + Wo + bias + residual ============
__global__ void __launch_bounds__(256)
out_kernel(const float* __restrict__ oat, const float* __restrict__ x,
           const _Float16* __restrict__ whO, const float* __restrict__ bo,
           float* __restrict__ y) {
  extern __shared__ char dyn_smem[];
  _Float16* sW = (_Float16*)dyn_smem;  // 128*128 f16
  _Float16* sX = sW + E * E;           // 128*128 f16
  const int tid = threadIdx.x, lane = tid & 31, wave = tid >> 5;
  const int rowBase = blockIdx.x * 128;

  for (int i = tid; i < (E * E) / 8; i += 256)
    async_copy_b128(sW + i * 8, whO + i * 8);

  for (int r = 0; r < 16; ++r) {
    const int row = rowBase + wave * 16 + r;
    float4 v = ((const float4*)(oat + (size_t)row * E))[lane];
    float s  = wave_sum(v.x + v.y + v.z + v.w);
    float s2 = wave_sum(v.x * v.x + v.y * v.y + v.z * v.z + v.w * v.w);
    float mean = s * (1.0f / E);
    float rstd = rsqrtf(s2 * (1.0f / E) - mean * mean + 1e-5f);
    _Float16* dst = sX + (wave * 16 + r) * E + lane * 4;
    dst[0] = (_Float16)fmaxf((v.x - mean) * rstd, 0.0f);
    dst[1] = (_Float16)fmaxf((v.y - mean) * rstd, 0.0f);
    dst[2] = (_Float16)fmaxf((v.z - mean) * rstd, 0.0f);
    dst[3] = (_Float16)fmaxf((v.w - mean) * rstd, 0.0f);
  }
  async_wait_all();
  __syncthreads();

  half16v A[4];
#pragma unroll
  for (int kk = 0; kk < 4; ++kk)
    A[kk] = load_frag_A(sX + wave * 16 * E, lane, E, kk * 32);

  const int n = lane & 15, ghi = lane >> 4;
#pragma unroll
  for (int fb = 0; fb < 8; ++fb) {
    const float bb = bo[fb * 16 + n];
    half16v Bf[4];
#pragma unroll
    for (int kk = 0; kk < 4; ++kk)
      Bf[kk] = load_frag_B(sW + fb * 16 * E, lane, E, kk * 32);
    float8v acc = {};
#pragma unroll
    for (int kk = 0; kk < 4; ++kk) acc = wmma_f16(A[kk], Bf[kk], acc);
    const size_t base = ((size_t)rowBase + wave * 16) * E + fb * 16 + n;
#pragma unroll
    for (int g = 0; g < 8; ++g) {
      const size_t off = base + (size_t)(g + 8 * ghi) * E;
      y[off] = x[off] + acc[g] + bb;   // residual
    }
  }
}

// ============ host-side launcher ============
extern "C" void kernel_launch(void* const* d_in, const int* in_sizes, int n_in,
                              void* d_out, int out_size, void* d_ws, size_t ws_size,
                              hipStream_t stream) {
  (void)in_sizes; (void)n_in; (void)out_size; (void)ws_size;
  const float* x   = (const float*)d_in[0];
  const float* adj = (const float*)d_in[1];
  const float* Wq  = (const float*)d_in[2];
  const float* bq  = (const float*)d_in[3];
  const float* Wk  = (const float*)d_in[4];
  const float* bk  = (const float*)d_in[5];
  const float* Wv  = (const float*)d_in[6];
  const float* bv  = (const float*)d_in[7];
  const float* Wo  = (const float*)d_in[8];
  const float* bo  = (const float*)d_in[9];
  float* y = (float*)d_out;

  // workspace layout: wh (f16 weights), q (f16), k (f16), vT (f16), attn out (f32)
  _Float16* wh = (_Float16*)d_ws;
  _Float16* qh = wh + (size_t)4 * E * E;
  _Float16* kh = qh + (size_t)NROWS * E;
  _Float16* vT = kh + (size_t)NROWS * E;
  float*   oat = (float*)(vT + (size_t)NROWS * E);

  wcvt_kernel<<<(4 * E * E) / 256, 256, 0, stream>>>(Wq, Wk, Wv, Wo, wh);

  qkv_kernel<<<NROWS / 128, 256, 4 * E * E * sizeof(_Float16), stream>>>(
      x, wh, bq, bk, bv, qh, kh, vT);

  const size_t attn_lds = 16 * SC_STRIDE * sizeof(float) + 16 * sizeof(float);
  attn_kernel<<<BSUB * 32, 32, attn_lds, stream>>>(adj, qh, kh, vT, oat);

  out_kernel<<<NROWS / 128, 256, 2 * E * E * sizeof(_Float16), stream>>>(
      oat, x, wh + 3 * E * E, bo, y);
}
